// EulerFlexRNNCell_58600533787375
// MI455X (gfx1250) — matrix operand order repaired and checked by reading, not compiled, not run on hardware
//
#include <hip/hip_runtime.h>
#include <hip/hip_bf16.h>

typedef __attribute__((ext_vector_type(16))) _Float16 v16h;
typedef __attribute__((ext_vector_type(8)))  float    v8f;

namespace {
constexpr int kB     = 1024;   // trajectories
constexpr int kD     = 10;     // complex state dim
constexpr int kM     = 2;      // noise channels
constexpr int kH     = 256;    // hidden width per net
constexpr int kT     = 128;    // output time points
constexpr int kSteps = 127;    // Euler steps
constexpr int kBT    = 16;     // trajectories per workgroup (= WMMA M)
constexpr double kDt = 1.0 / 128.0;
constexpr int kHidW      = 4 * kH;        // 1024 concat hidden cols
constexpr int kHidStride = kHidW + 4;     // pad (halves) to avoid bank conflicts
constexpr int kFStride   = 36;            // padded 32-col f tile
constexpr int kOutStride = 100;           // padded 96-col layer2 out
}

union FragH { unsigned int u[8]; _Float16 h[16]; v16h v; };
union FragF { float f[8]; v8f v; };

// Branchless saturating tanh on native v_exp_f32 / v_rcp_f32:
// tanh(x) = 1 - 2/(exp(2x)+1).  |x| large+ -> exp=inf -> rcp=0 -> +1;
// |x| large- -> exp=0 -> rcp(1)=1 -> -1.  No divergence, ~6 ops.
__device__ inline float fast_tanh(float x) {
  float e = __builtin_amdgcn_exp2f(x * 2.8853900817779268f);  // 2*log2(e)
  float r = __builtin_amdgcn_rcpf(e + 1.0f);
  return __builtin_fmaf(-2.0f, r, 1.0f);
}

// Build a 32x16 (KxN) f16 B-fragment from a row-major f64 weight matrix.
// Contiguous k order chosen here, same as A: lane<16 -> K=kBase..kBase+15 at N=lane; lane>=16 -> K+16.
__device__ inline v16h load_bfrag_f64(const double* W, int ldn, int nCol, int nMax,
                                      int kBase, int kMax) {
  FragH fr;
#pragma unroll
  for (int e = 0; e < 16; ++e) {
    int k = kBase + e;
    float val = 0.0f;
    if (nCol < nMax && k < kMax) val = (float)W[(long)k * ldn + nCol];
    fr.h[e] = (_Float16)val;
  }
  return fr.v;
}

// Build a 16x32 (MxK) f16 A-fragment from an LDS row (this lane's M row).
// VGPR v holds K pair (kBase + 2v).
__device__ inline v16h load_afrag_lds(const _Float16* row, int kBase) {
  FragH fr;
#pragma unroll
  for (int j = 0; j < 8; ++j) {
    int off = kBase + 2 * j;
    fr.u[j] = *(const unsigned int*)(row + off);
  }
  return fr.v;
}

__global__ __launch_bounds__(256) void euler_sde_wmma_kernel(
    const double* __restrict__ rho0,  const double* __restrict__ params,
    const double* __restrict__ wvec,  const double* __restrict__ Amat,
    const double* __restrict__ Bten,
    const double* __restrict__ Wa1r, const double* __restrict__ ba1r,
    const double* __restrict__ Wa2r, const double* __restrict__ ba2r,
    const double* __restrict__ Wa1i, const double* __restrict__ ba1i,
    const double* __restrict__ Wa2i, const double* __restrict__ ba2i,
    const double* __restrict__ Wb1r, const double* __restrict__ bb1r,
    const double* __restrict__ Wb2r, const double* __restrict__ bb2r,
    const double* __restrict__ Wb1i, const double* __restrict__ bb1i,
    const double* __restrict__ Wb2i, const double* __restrict__ bb2i,
    double* __restrict__ out) {
  __shared__ _Float16 sF[kBT][kFStride];        // f tile [16 x 32] (K padded 24->32)
  __shared__ _Float16 sHid[kBT][kHidStride];    // tanh activations, 4 nets concat
  __shared__ float    sOut2[kBT][kOutStride];   // layer-2 results (96 valid cols)
  __shared__ float    sB1[kHidW];               // layer-1 bias concat
  __shared__ double   sXr[kBT][kD], sXi[kBT][kD];
  __shared__ double   sPar[kBT][4];
  __shared__ double   sAr[kD][kD], sAi[kD][kD];
  __shared__ double   sBtr[kD][kD][kM], sBti[kD][kD][kM];

  const int tid  = threadIdx.x;
  const int lane = tid & 31;
  const int wv   = tid >> 5;
  const int b0   = blockIdx.x * kBT;

  // ---- stage state / params / physics tensors into LDS ----
  if (tid < kBT * kD) {
    int b = tid / kD, idx = tid % kD;
    int i, j;
    if (idx < 4)      { i = 0; j = idx; }
    else if (idx < 7) { i = 1; j = idx - 3; }
    else if (idx < 9) { i = 2; j = idx - 5; }
    else              { i = 3; j = 3; }
    long g = ((long)(b0 + b) * 16 + i * 4 + j) * 2;
    sXr[b][idx] = rho0[g];
    sXi[b][idx] = rho0[g + 1];
  }
  if (tid < kBT * 4) {
    int b = tid >> 2, p = tid & 3;
    sPar[b][p] = params[(long)(b0 + b) * 4 + p];
  }
  for (int c = tid; c < kHidW; c += 256) {
    int net = c >> 8, ci = c & 255;
    const double* bp = (net == 0) ? ba1r : (net == 1) ? ba1i : (net == 2) ? bb1r : bb1i;
    sB1[c] = (float)bp[ci];
  }
  if (tid < kD * kD) {
    int d = tid / kD, k = tid % kD;
    sAr[d][k] = Amat[(d * kD + k) * 2];
    sAi[d][k] = Amat[(d * kD + k) * 2 + 1];
  }
  if (tid < kD * kD * kM) {
    int k = tid / (kD * kM), r = tid % (kD * kM);
    int d = r / kM, m = r % kM;
    sBtr[k][d][m] = Bten[((k * kD + d) * kM + m) * 2];
    sBti[k][d][m] = Bten[((k * kD + d) * kM + m) * 2 + 1];
  }

  // ---- convert weights f64 -> f16 B-fragments held in VGPRs (once) ----
  const int nl  = lane & 15;
  const int kbB = (lane < 16) ? 0 : 16;   // B-fragment K base within 32-window
  const int kbA = (lane < 16) ? 0 : 16;   // A-fragment K base within 32-window

  v16h w1f[8];  // each wave: 8 N-tiles of the [32 x 1024] concat layer-1 weights
#pragma unroll
  for (int j = 0; j < 8; ++j) {
    int tn  = wv * 8 + j;        // tile 0..63, 16 tiles per net
    int net = tn >> 4;
    const double* W1 = (net == 0) ? Wa1r : (net == 1) ? Wa1i : (net == 2) ? Wb1r : Wb1i;
    int nCol = (tn & 15) * 16 + nl;
    w1f[j] = load_bfrag_f64(W1, kH, nCol, kH, kbB, 24);  // K valid 0..23
  }
  // layer-2 tiles: w0 a_r | w1 a_i | w2/w3 b_r lo/hi | w4/w5 b_i lo/hi
  const double* W2 = (wv == 0) ? Wa2r : (wv == 1) ? Wa2i : (wv <= 3) ? Wb2r : Wb2i;
  const int fo2   = (wv < 2) ? kD : kD * kM;
  const int noff2 = ((wv == 3) || (wv == 5)) ? 16 : 0;
  const int net2  = (wv < 2) ? wv : (wv < 4) ? 2 : 3;
  v16h w2f[8];
#pragma unroll
  for (int kk = 0; kk < 8; ++kk) {
    if (wv < 6) w2f[kk] = load_bfrag_f64(W2, fo2, noff2 + nl, fo2, kk * 32 + kbB, kH);
    else        w2f[kk] = load_bfrag_f64(Wa2r, kD, 16, kD, 0, 0);  // zeros
  }
  __syncthreads();

#pragma unroll 1
  for (int t = 0; t < kT; ++t) {
    // ---- emit rho(t) = unwrap(x) as complex128, 16 cells per trajectory ----
    {
      int bl = tid >> 4, cell = tid & 15;
      int i = cell >> 2, j = cell & 3;
      int ii = (i <= j) ? i : j, jj = (i <= j) ? j : i;
      int idx = 4 * ii - (ii * (ii - 1)) / 2 + (jj - ii);
      double re = sXr[bl][idx];
      double im = (i <= j) ? sXi[bl][idx] : -sXi[bl][idx];
      long g = (((long)(b0 + bl) * kT + t) * 16 + cell) * 2;
      out[g]     = re;
      out[g + 1] = im;
    }
    if (t == kSteps) break;

    // ---- build f = [Re x, Im x, params, 0pad] tile in f16 ----
    for (int e = tid; e < kBT * 32; e += 256) {
      int b = e >> 5, c = e & 31;
      float v;
      if (c < kD)              v = (float)sXr[b][c];
      else if (c < 2 * kD)     v = (float)sXi[b][c - kD];
      else if (c < 2 * kD + 4) v = (float)sPar[b][c - 2 * kD];
      else                     v = 0.0f;
      sF[b][c] = (_Float16)v;
    }
    __syncthreads();

    // ---- GEMM1: [16,32] x [32,1024]; bias pre-loaded into C (per-lane col) ----
    {
      v16h af = load_afrag_lds(&sF[nl][0], kbA);
      v8f acc[8];
#pragma unroll
      for (int j = 0; j < 8; ++j) {
        float bias = sB1[(wv * 8 + j) * 16 + nl];
        v8f cinit = {bias, bias, bias, bias, bias, bias, bias, bias};
        acc[j] = __builtin_amdgcn_wmma_f32_16x16x32_f16(false, af, false, w1f[j],
                                                        (short)0, cinit, false, false);
      }
      int mr0 = (lane >> 4) * 8;
#pragma unroll
      for (int j = 0; j < 8; ++j) {
        int ncol = (wv * 8 + j) * 16 + nl;
        FragF fr; fr.v = acc[j];
#pragma unroll
        for (int r = 0; r < 8; ++r)
          sHid[mr0 + r][ncol] = (_Float16)fast_tanh(fr.f[r]);
      }
    }
    __syncthreads();

    // ---- GEMM2: [16,256] x [256,16] per tile, 8 K-steps (waves 0..5) ----
    if (wv < 6) {
      v8f c2 = {0.f, 0.f, 0.f, 0.f, 0.f, 0.f, 0.f, 0.f};
#pragma unroll
      for (int kk = 0; kk < 8; ++kk) {
        v16h a2 = load_afrag_lds(&sHid[nl][0], net2 * kH + kk * 32 + kbA);
        c2 = __builtin_amdgcn_wmma_f32_16x16x32_f16(false, a2, false, w2f[kk],
                                                    (short)0, c2, false, false);
      }
      int mr0 = (lane >> 4) * 8;
      FragF fr; fr.v = c2;
#pragma unroll
      for (int r = 0; r < 8; ++r)
        sOut2[mr0 + r][wv * 16 + nl] = fr.f[r];
    }
    __syncthreads();

    // ---- fp64 complex Euler update: one thread per (trajectory, d) ----
    double nxr = 0.0, nxi = 0.0;
    const int ub = tid / kD, ud = tid % kD;
    if (tid < kBT * kD) {
      double ar = 0.0, ai = 0.0;
#pragma unroll
      for (int k = 0; k < kD; ++k) {
        double xr = sXr[ub][k], xi = sXi[ub][k];
        ar += sAr[ud][k] * xr - sAi[ud][k] * xi;
        ai += sAr[ud][k] * xi + sAi[ud][k] * xr;
      }
      ar += (double)sOut2[ub][ud]      + ba2r[ud];
      ai += (double)sOut2[ub][16 + ud] + ba2i[ud];
      nxr = sXr[ub][ud] + ar * kDt;
      nxi = sXi[ub][ud] + ai * kDt;
#pragma unroll
      for (int mch = 0; mch < kM; ++mch) {
        double br = 0.0, bi = 0.0;
#pragma unroll
        for (int k = 0; k < kD; ++k) {
          double xr = sXr[ub][k], xi = sXi[ub][k];
          br += sBtr[k][ud][mch] * xr - sBti[k][ud][mch] * xi;
          bi += sBtr[k][ud][mch] * xi + sBti[k][ud][mch] * xr;
        }
        int dm = ud * kM + mch;
        br += (double)sOut2[ub][32 + dm] + bb2r[dm];
        bi += (double)sOut2[ub][64 + dm] + bb2i[dm];
        double dw = wvec[((long)(b0 + ub) * kSteps + t) * kM + mch];
        nxr += br * dw;
        nxi += bi * dw;
      }
    }
    __syncthreads();
    if (tid < kBT * kD) {
      sXr[ub][ud] = nxr;
      sXi[ub][ud] = nxi;
    }
    __syncthreads();
  }
}

extern "C" void kernel_launch(void* const* d_in, const int* in_sizes, int n_in,
                              void* d_out, int out_size, void* d_ws, size_t ws_size,
                              hipStream_t stream) {
  (void)in_sizes; (void)n_in; (void)out_size; (void)d_ws; (void)ws_size;
  const double* p[21];
  for (int i = 0; i < 21; ++i) p[i] = (const double*)d_in[i];
  euler_sde_wmma_kernel<<<kB / kBT, 256, 0, stream>>>(
      p[0], p[1], p[2], p[3], p[4],
      p[5], p[6], p[7], p[8],
      p[9], p[10], p[11], p[12],
      p[13], p[14], p[15], p[16],
      p[17], p[18], p[19], p[20],
      (double*)d_out);
}
